// InterestEvolvingLayer_1941325217798
// MI455X (gfx1250) — matrix operand order repaired and checked by reading, not compiled, not run on hardware
//
#include <hip/hip_runtime.h>
#include <hip/hip_bf16.h>

// Problem sizes (fixed by the reference)
#define B_SZ 4096
#define T_SZ 200
#define D_SZ 128
#define H_SZ 128
#define A_SZ 64

typedef __attribute__((ext_vector_type(16))) __bf16 v16bf;
typedef __attribute__((ext_vector_type(8)))  float v8f;

union Frag { unsigned int u[8]; v16bf v; };

__device__ __forceinline__ unsigned int f2bf(float f) {        // RNE (one-shot paths)
  unsigned int u = __float_as_uint(f);
  return (u + 0x7FFFu + ((u >> 16) & 1u)) >> 16;
}
// Truncate-pack two fp32 -> packed bf16 in ONE v_perm_b32.
__device__ __forceinline__ unsigned int pack2(float lo, float hi) {
  return __builtin_amdgcn_perm(__float_as_uint(hi), __float_as_uint(lo), 0x07060302u);
}
__device__ __forceinline__ unsigned short bftrunc(float f) {
  return (unsigned short)(__float_as_uint(f) >> 16);
}
__device__ __forceinline__ float sigmoidf(float x) {
  return 1.0f / (1.0f + __expf(-x));
}

// Workspace layout (bytes)
static constexpr size_t OFF_WA1 = 0;                        // 64*256 bf16 = 32768
static constexpr size_t OFF_WRX = 32768;                    // 128*128 bf16 each
static constexpr size_t OFF_WRH = 65536;
static constexpr size_t OFF_WZX = 98304;
static constexpr size_t OFF_WZH = 131072;
static constexpr size_t OFF_WHX = 163840;
static constexpr size_t OFF_WHH = 196608;
static constexpr size_t OFF_RX  = 229376;                   // 4096*128 f32 = 2 MiB
static constexpr size_t OFF_ZX  = OFF_RX + 2097152;
static constexpr size_t OFF_CX  = OFF_ZX + 2097152;
static constexpr size_t OFF_ATT = OFF_CX + 2097152;         // 4096*200 f32

static constexpr int XP_PITCH = 132;   // floats
static constexpr int H_PITCH  = 132;   // floats (fp32 h)
static constexpr int HB_PITCH = 136;   // ushorts (bf16 shadows; 272B rows, 16B aligned)
static constexpr int A_PITCH  = 201;   // floats (bank spread)

// ---------------------------------------------------------------------------
// Kernel 0: convert weights fp32 -> bf16 (RNE) into workspace
// ---------------------------------------------------------------------------
__global__ void convert_weights(const float* __restrict__ Wa1,
                                const float* __restrict__ Wr,
                                const float* __restrict__ Wz,
                                const float* __restrict__ Wh,
                                unsigned short* __restrict__ wa1,
                                unsigned short* __restrict__ wrx, unsigned short* __restrict__ wrh,
                                unsigned short* __restrict__ wzx, unsigned short* __restrict__ wzh,
                                unsigned short* __restrict__ whx, unsigned short* __restrict__ whh) {
  int i = blockIdx.x * blockDim.x + threadIdx.x;
  if (i < A_SZ * (H_SZ + D_SZ)) wa1[i] = (unsigned short)f2bf(Wa1[i]);
  if (i < H_SZ * H_SZ) {
    int j = i >> 7, k = i & 127;
    wrx[i] = (unsigned short)f2bf(Wr[j * 256 + k]);
    wrh[i] = (unsigned short)f2bf(Wr[j * 256 + 128 + k]);
    wzx[i] = (unsigned short)f2bf(Wz[j * 256 + k]);
    wzh[i] = (unsigned short)f2bf(Wz[j * 256 + 128 + k]);
    whx[i] = (unsigned short)f2bf(Wh[j * 256 + k]);
    whh[i] = (unsigned short)f2bf(Wh[j * 256 + 128 + k]);
  }
}

// Build an A-fragment K-chunk (32 wide) from a row-major fp32 row.
__device__ __forceinline__ Frag load_a_frag(const float* __restrict__ row,
                                            int kbase, int half) {
  Frag f;
  const float* p = row + kbase + (half ? 8 : 0);
  float4 f0 = *(const float4*)(p);
  float4 f1 = *(const float4*)(p + 4);
  float4 f2 = *(const float4*)(p + 16);
  float4 f3 = *(const float4*)(p + 20);
  f.u[0] = pack2(f0.x, f0.y); f.u[1] = pack2(f0.z, f0.w);
  f.u[2] = pack2(f1.x, f1.y); f.u[3] = pack2(f1.z, f1.w);
  f.u[4] = pack2(f2.x, f2.y); f.u[5] = pack2(f2.z, f2.w);
  f.u[6] = pack2(f3.x, f3.y); f.u[7] = pack2(f3.z, f3.w);
  return f;
}
// Load a B-fragment (32x16 bf16) from a row-major bf16 weight row block.
__device__ __forceinline__ Frag load_b_frag(const unsigned short* __restrict__ w,
                                            int jrow, int rowlen, int kc, int half) {
  Frag f;
  const unsigned short* p = w + jrow * rowlen + kc * 32 + (half ? 8 : 0);
  *(uint4*)&f.u[0] = *(const uint4*)p;
  *(uint4*)&f.u[4] = *(const uint4*)(p + 16);
  return f;
}

// ---------------------------------------------------------------------------
// Kernel 1: x-projections r_x/z_x/c_x = target @ W_x^T + bias  (WMMA)
// ---------------------------------------------------------------------------
__global__ void xproj_kernel(const float* __restrict__ target,
                             const unsigned short* __restrict__ wrx,
                             const unsigned short* __restrict__ wzx,
                             const unsigned short* __restrict__ whx,
                             const float* __restrict__ br, const float* __restrict__ bz,
                             const float* __restrict__ bh,
                             float* __restrict__ rx, float* __restrict__ zx,
                             float* __restrict__ cx) {
  const int wave = threadIdx.x >> 5;
  const int lane = threadIdx.x & 31;
  const int half = lane >> 4, ln = lane & 15;
  const int tile = blockIdx.x * 4 + wave;
  const int row0 = tile * 16;

  Frag aF[4];
  const float* trow = target + (row0 + ln) * D_SZ;
#pragma unroll
  for (int kc = 0; kc < 4; ++kc) aF[kc] = load_a_frag(trow, kc * 32, half);

  const unsigned short* wx[3]  = { wrx, wzx, whx };
  const float*          bia[3] = { br, bz, bh };
  float*                out[3] = { rx, zx, cx };
#pragma unroll
  for (int g = 0; g < 3; ++g) {
#pragma unroll
    for (int nc = 0; nc < 8; ++nc) {
      int j = nc * 16 + ln;
      float bj = bia[g][j];
      Frag bF[4];
#pragma unroll
      for (int kc = 0; kc < 4; ++kc) bF[kc] = load_b_frag(wx[g], j, H_SZ, kc, half);
      v8f acc;
#pragma unroll
      for (int v = 0; v < 8; ++v) acc[v] = bj;
#pragma unroll
      for (int kc = 0; kc < 4; ++kc)
        acc = __builtin_amdgcn_wmma_f32_16x16x32_bf16(false, aF[kc].v, false, bF[kc].v,
                                                      (short)0, acc, false, false);
#pragma unroll
      for (int v = 0; v < 8; ++v) {
        int m = v + (half ? 8 : 0);
        out[g][(row0 + m) * H_SZ + j] = acc[v];
      }
    }
  }
}

// ---------------------------------------------------------------------------
// Kernel 2: attention  att = relu([seq,tgt] @ Wa1^T + ba1) @ Wa2^T + ba2
// ---------------------------------------------------------------------------
__global__ void attn_kernel(const float* __restrict__ interest,
                            const float* __restrict__ target,
                            const unsigned short* __restrict__ wa1,
                            const float* __restrict__ ba1,
                            const float* __restrict__ Wa2,
                            const float* __restrict__ ba2,
                            float* __restrict__ att) {
  const int wave = threadIdx.x >> 5;
  const int lane = threadIdx.x & 31;
  const int half = lane >> 4, ln = lane & 15;
  const long tile = (long)blockIdx.x * 4 + wave;   // 51200 tiles
  const long row0 = tile * 16;
  const long r = row0 + ln;
  const int b = (int)(r / T_SZ);
  const int t = (int)(r % T_SZ);
  const float* irow = interest + ((long)b * T_SZ + t) * H_SZ;
  const float* trow = target + (long)b * D_SZ;

  // hoisted second-layer weights/biases for this lane's 4 columns
  float b1a[4], w2a[4];
#pragma unroll
  for (int nc = 0; nc < 4; ++nc) {
    b1a[nc] = ba1[nc * 16 + ln];
    w2a[nc] = Wa2[nc * 16 + ln];
  }

  Frag aF[8];
#pragma unroll
  for (int kc = 0; kc < 4; ++kc) aF[kc] = load_a_frag(irow, kc * 32, half);
#pragma unroll
  for (int kc = 0; kc < 4; ++kc) aF[kc + 4] = load_a_frag(trow, kc * 32, half);

  float p[8];
#pragma unroll
  for (int v = 0; v < 8; ++v) p[v] = 0.0f;

#pragma unroll
  for (int nc = 0; nc < 4; ++nc) {
    int a = nc * 16 + ln;
    Frag bF[8];
#pragma unroll
    for (int kc = 0; kc < 8; ++kc) bF[kc] = load_b_frag(wa1, a, 256, kc, half);
    v8f acc;
#pragma unroll
    for (int v = 0; v < 8; ++v) acc[v] = b1a[nc];
#pragma unroll
    for (int kc = 0; kc < 8; ++kc)
      acc = __builtin_amdgcn_wmma_f32_16x16x32_bf16(false, aF[kc].v, false, bF[kc].v,
                                                    (short)0, acc, false, false);
#pragma unroll
    for (int v = 0; v < 8; ++v) {
      float hrelu = acc[v] > 0.0f ? acc[v] : 0.0f;
      p[v] += hrelu * w2a[nc];
    }
  }
#pragma unroll
  for (int off = 1; off < 16; off <<= 1)
#pragma unroll
    for (int v = 0; v < 8; ++v) p[v] += __shfl_xor(p[v], off, 32);

  if (ln == 0) {
    float b2 = ba2[0];
#pragma unroll
    for (int v = 0; v < 8; ++v) {
      long m = v + (half ? 8 : 0);
      att[row0 + m] = p[v] + b2;
    }
  }
}

// ---------------------------------------------------------------------------
// Kernel 3: AUGRU recurrence. Block = 8 waves (256 thr) on one 16-row tile.
// Each wave owns ONE N-tile; its 12 weight B-fragments (3 gates x 4 K-chunks)
// live in VGPRs for the whole kernel. h: fp32 + bf16 shadow in LDS.
// hc = tanh(c_x + (r*h) @ Wh_h^T).
// ---------------------------------------------------------------------------
__global__ void __launch_bounds__(256) augru_kernel(
    const int* __restrict__ seq_len,
    const unsigned short* __restrict__ wrh,
    const unsigned short* __restrict__ wzh,
    const unsigned short* __restrict__ whh,
    const float* __restrict__ rx, const float* __restrict__ zx,
    const float* __restrict__ cx,
    const float* __restrict__ att,
    float* __restrict__ out) {
  __shared__ float          xp[3][16][XP_PITCH];        // g: 0=zx 1=rx 2=cx
  __shared__ float          hL[16][H_PITCH];
  __shared__ unsigned short hbf[16][HB_PITCH];
  __shared__ unsigned short rhbf[16][HB_PITCH];
  __shared__ float          aL[16][A_PITCH];

  const int tid  = threadIdx.x;
  const int wave = tid >> 5;          // 0..7 == N-tile index
  const int lane = tid & 31;
  const int half = lane >> 4, ln = lane & 15;
  const int rb = blockIdx.x * 16;
  const int j  = wave * 16 + ln;      // this lane's output column

  // ---- persistent weight B-fragments in registers: [gate][kc], g:0=z 1=r 2=h
  Frag wF[3][4];
#pragma unroll
  for (int kc = 0; kc < 4; ++kc) {
    wF[0][kc] = load_b_frag(wzh, j, H_SZ, kc, half);
    wF[1][kc] = load_b_frag(wrh, j, H_SZ, kc, half);
    wF[2][kc] = load_b_frag(whh, j, H_SZ, kc, half);
  }

  // ---- stage per-tile data in LDS ----
  for (int idx = tid; idx < 3 * 16 * H_SZ; idx += 256) {
    int g = idx >> 11;
    int r_ = idx & 2047;
    int m = r_ >> 7, jj = r_ & 127;
    const float* src = (g == 0) ? zx : (g == 1) ? rx : cx;
    xp[g][m][jj] = src[(rb + m) * H_SZ + jj];
  }
  for (int idx = tid; idx < 16 * H_PITCH; idx += 256) ((float*)hL)[idx] = 0.0f;
  for (int idx = tid; idx < 16 * HB_PITCH; idx += 256) {
    ((unsigned short*)hbf)[idx] = 0;
    ((unsigned short*)rhbf)[idx] = 0;
  }
  for (int idx = tid; idx < 16 * T_SZ; idx += 256) {
    int m = idx / T_SZ, t = idx % T_SZ;
    aL[m][t] = att[(long)(rb + m) * T_SZ + t];
  }
  __syncthreads();

  int slv[8];
#pragma unroll
  for (int v = 0; v < 8; ++v) {
    int m = v + (half ? 8 : 0);
    int s = seq_len[rb + m];
    slv[v] = s < 1 ? 1 : s;
  }

  float zreg[8];

  for (int t = 0; t < T_SZ; ++t) {
    // phase 1: A-fragments of h from the bf16 shadow (b128 LDS loads)
    Frag hA[4];
#pragma unroll
    for (int kc = 0; kc < 4; ++kc) {
      const unsigned short* ph = &hbf[ln][0] + kc * 32 + (half ? 8 : 0);
      *(uint4*)&hA[kc].u[0] = *(const uint4*)ph;
      *(uint4*)&hA[kc].u[4] = *(const uint4*)(ph + 16);
    }

    // phase 2: z, r on this wave's N-tile; stage bf16(r*h)
    v8f zacc, racc;
#pragma unroll
    for (int v = 0; v < 8; ++v) {
      int m = v + (half ? 8 : 0);
      zacc[v] = xp[0][m][j];
      racc[v] = xp[1][m][j];
    }
#pragma unroll
    for (int kc = 0; kc < 4; ++kc) {
      zacc = __builtin_amdgcn_wmma_f32_16x16x32_bf16(false, hA[kc].v, false, wF[0][kc].v,
                                                     (short)0, zacc, false, false);
      racc = __builtin_amdgcn_wmma_f32_16x16x32_bf16(false, hA[kc].v, false, wF[1][kc].v,
                                                     (short)0, racc, false, false);
    }
#pragma unroll
    for (int v = 0; v < 8; ++v) {
      int m = v + (half ? 8 : 0);
      zreg[v] = sigmoidf(zacc[v]);
      float rr = sigmoidf(racc[v]);
      rhbf[m][j] = bftrunc(rr * hL[m][j]);
    }
    __syncthreads();

    // phase 3: A-fragments of (r*h)
    Frag rhA[4];
#pragma unroll
    for (int kc = 0; kc < 4; ++kc) {
      const unsigned short* pr = &rhbf[ln][0] + kc * 32 + (half ? 8 : 0);
      *(uint4*)&rhA[kc].u[0] = *(const uint4*)pr;
      *(uint4*)&rhA[kc].u[4] = *(const uint4*)(pr + 16);
    }

    float a_v[8], m_v[8];
#pragma unroll
    for (int v = 0; v < 8; ++v) {
      int m = v + (half ? 8 : 0);
      a_v[v] = aL[m][t];
      m_v[v] = (t < slv[v]) ? 1.0f : 0.0f;
    }

    // phase 4: hc and gated h update
    v8f cacc;
#pragma unroll
    for (int v = 0; v < 8; ++v) {
      int m = v + (half ? 8 : 0);
      cacc[v] = xp[2][m][j];
    }
#pragma unroll
    for (int kc = 0; kc < 4; ++kc)
      cacc = __builtin_amdgcn_wmma_f32_16x16x32_bf16(false, rhA[kc].v, false, wF[2][kc].v,
                                                     (short)0, cacc, false, false);
#pragma unroll
    for (int v = 0; v < 8; ++v) {
      int m = v + (half ? 8 : 0);
      float hc = tanhf(cacc[v]);
      float az = a_v[v] * zreg[v];
      float hold = hL[m][j];
      float hn = (1.0f - az) * hold + az * hc;
      float hupd = m_v[v] * hn + (1.0f - m_v[v]) * hold;
      hL[m][j] = hupd;
      hbf[m][j] = bftrunc(hupd);
    }
    __syncthreads();
  }

  for (int idx = tid; idx < 16 * H_SZ; idx += 256) {
    int m = idx >> 7, jj = idx & 127;
    out[(long)(rb + m) * H_SZ + jj] = hL[m][jj];
  }
}

// ---------------------------------------------------------------------------
extern "C" void kernel_launch(void* const* d_in, const int* in_sizes, int n_in,
                              void* d_out, int out_size, void* d_ws, size_t ws_size,
                              hipStream_t stream) {
  const float* interest = (const float*)d_in[0];
  const float* target   = (const float*)d_in[1];
  const int*   seq_len  = (const int*)d_in[2];
  const float* Wr  = (const float*)d_in[3];
  const float* br  = (const float*)d_in[4];
  const float* Wz  = (const float*)d_in[5];
  const float* bz  = (const float*)d_in[6];
  const float* Wh  = (const float*)d_in[7];
  const float* bh  = (const float*)d_in[8];
  const float* Wa1 = (const float*)d_in[9];
  const float* ba1 = (const float*)d_in[10];
  const float* Wa2 = (const float*)d_in[11];
  const float* ba2 = (const float*)d_in[12];
  float* out = (float*)d_out;

  char* ws = (char*)d_ws;
  unsigned short* wa1 = (unsigned short*)(ws + OFF_WA1);
  unsigned short* wrx = (unsigned short*)(ws + OFF_WRX);
  unsigned short* wrh = (unsigned short*)(ws + OFF_WRH);
  unsigned short* wzx = (unsigned short*)(ws + OFF_WZX);
  unsigned short* wzh = (unsigned short*)(ws + OFF_WZH);
  unsigned short* whx = (unsigned short*)(ws + OFF_WHX);
  unsigned short* whh = (unsigned short*)(ws + OFF_WHH);
  float* rxp = (float*)(ws + OFF_RX);
  float* zxp = (float*)(ws + OFF_ZX);
  float* cxp = (float*)(ws + OFF_CX);
  float* att = (float*)(ws + OFF_ATT);

  convert_weights<<<64, 256, 0, stream>>>(Wa1, Wr, Wz, Wh, wa1,
                                          wrx, wrh, wzx, wzh, whx, whh);
  xproj_kernel<<<64, 128, 0, stream>>>(target, wrx, wzx, whx, br, bz, bh,
                                       rxp, zxp, cxp);
  attn_kernel<<<12800, 128, 0, stream>>>(interest, target, wa1, ba1, Wa2, ba2, att);
  augru_kernel<<<256, 256, 0, stream>>>(seq_len, wrh, wzh, whh, rxp, zxp, cxp,
                                        att, out);
}
